// MultiheadSelfAttention_46385646797427
// MI455X (gfx1250) — matrix operand order repaired and checked by reading, not compiled, not run on hardware
//
#include <hip/hip_runtime.h>
#include <hip/hip_bf16.h>

#define S_LEN  1024
#define M_LEN  1024
#define T_LEN  2048
#define BATCH  4
#define DMODEL 1024
#define NHEAD  16
#define DHEAD  64
#define QKV3   3072
#define SCALE  0.125f
#define EPS    1e-5f
#define NEGBIG (-1.0e30f)

#if defined(__has_builtin)
#if __has_builtin(__builtin_amdgcn_global_load_async_to_lds_b128)
#define HAVE_ASYNC_COPY 1
#endif
#endif

typedef __attribute__((ext_vector_type(8)))  __bf16 v8bf;
typedef __attribute__((ext_vector_type(16))) __bf16 v16bf;
typedef __attribute__((ext_vector_type(8)))  float  v8f;
typedef __attribute__((ext_vector_type(4)))  int    v4i;
typedef __attribute__((address_space(1)))    v4i    gv4i;   // global int4
typedef __attribute__((address_space(3)))    v4i    lv4i;   // LDS int4

static __device__ __forceinline__ v16bf frag16(const __bf16* lo, const __bf16* hi) {
  v8bf a = *(const v8bf*)lo;
  v8bf b = *(const v8bf*)hi;
  return __builtin_shufflevector(a, b, 0,1,2,3,4,5,6,7,8,9,10,11,12,13,14,15);
}

static __device__ __forceinline__ v8f wmma_bf16(v16bf a, v16bf b, v8f c) {
  return __builtin_amdgcn_wmma_f32_16x16x32_bf16(false, a, false, b, (short)0, c, false, false);
}

static __device__ __forceinline__ void wait_async0() {
#if defined(__has_builtin)
#if __has_builtin(__builtin_amdgcn_s_wait_asynccnt)
  __builtin_amdgcn_s_wait_asynccnt(0);
  return;
#endif
#endif
  asm volatile("s_wait_asynccnt 0" ::: "memory");
}

// Build c = [memory ; x + pos] in bf16.  Flat index n = (t*B+b)*DM + d.
__global__ void pack_c_kernel(const float* __restrict__ x, const float* __restrict__ pos,
                              const float* __restrict__ mem, __bf16* __restrict__ cb) {
  size_t n = (size_t)blockIdx.x * blockDim.x + threadIdx.x;
  int d = (int)(n & (DMODEL - 1));
  int r = (int)(n >> 10);            // t*B + b
  int t = r >> 2;                    // B == 4
  int b = r & 3;
  float v;
  if (t < M_LEN) {
    v = mem[n];                      // same flat layout for t < M
  } else {
    int s = t - M_LEN;
    v = x[(((size_t)s * BATCH + b) << 10) + d] + pos[((size_t)s << 10) + d];
  }
  cb[n] = (__bf16)v;
}

__global__ void f32_to_bf16_kernel(const float* __restrict__ in, __bf16* __restrict__ out, int n) {
  int i = blockIdx.x * blockDim.x + threadIdx.x;
  if (i < n) out[i] = (__bf16)in[i];
}

// C[M_,N] (f32 acc -> OT) = A[M_,K](bf16, row-major) * W[N,K](bf16, row-major)^T
// 8 waves/block; each wave owns a 32x64 output block (2x4 WMMA tiles, 8 WMMA/K-step).
// K-loop is software-pipelined: fragments for kb+32 are fetched before the WMMAs of kb.
template <typename OT>
__global__ void gemm_bf16_nt_kernel(const __bf16* __restrict__ A, const __bf16* __restrict__ W,
                                    OT* __restrict__ C, int K, int N) {
  int lane = threadIdx.x & 31;
  int wave = threadIdx.x >> 5;
  int half = lane >> 4;
  int ln   = lane & 15;
  int tm = blockIdx.y * 64  + (wave >> 2) * 32;   // 2 wave-rows
  int tn = blockIdx.x * 256 + (wave & 3) * 64;    // 4 wave-cols
  const __bf16* a0 = A + (size_t)(tm + ln)      * K;
  const __bf16* a1 = A + (size_t)(tm + 16 + ln) * K;
  const __bf16* w0 = W + (size_t)(tn + ln)      * K;
  const __bf16* w1 = W + (size_t)(tn + 16 + ln) * K;
  const __bf16* w2 = W + (size_t)(tn + 32 + ln) * K;
  const __bf16* w3 = W + (size_t)(tn + 48 + ln) * K;
  const int alo = half ? 8 : 0,  ahi = half ? 24 : 16;   // A 16x32 lane layout
  const int blo = half ? 8 : 0,  bhi = half ? 24 : 16;   // B 32x16 lane layout

  v8f acc[2][4];
  #pragma unroll
  for (int i = 0; i < 2; ++i)
    #pragma unroll
    for (int j = 0; j < 4; ++j) acc[i][j] = (v8f){};

  v16bf Af[2], Bf[4];
  Af[0] = frag16(a0 + alo, a0 + ahi);
  Af[1] = frag16(a1 + alo, a1 + ahi);
  Bf[0] = frag16(w0 + blo, w0 + bhi);
  Bf[1] = frag16(w1 + blo, w1 + bhi);
  Bf[2] = frag16(w2 + blo, w2 + bhi);
  Bf[3] = frag16(w3 + blo, w3 + bhi);

  for (int kb = 32; kb < K; kb += 32) {
    // prefetch next K-step into a second register set
    v16bf nAf0 = frag16(a0 + kb + alo, a0 + kb + ahi);
    v16bf nAf1 = frag16(a1 + kb + alo, a1 + kb + ahi);
    v16bf nBf0 = frag16(w0 + kb + blo, w0 + kb + bhi);
    v16bf nBf1 = frag16(w1 + kb + blo, w1 + kb + bhi);
    v16bf nBf2 = frag16(w2 + kb + blo, w2 + kb + bhi);
    v16bf nBf3 = frag16(w3 + kb + blo, w3 + kb + bhi);
    #pragma unroll
    for (int i = 0; i < 2; ++i) {
      acc[i][0] = wmma_bf16(Af[i], Bf[0], acc[i][0]);
      acc[i][1] = wmma_bf16(Af[i], Bf[1], acc[i][1]);
      acc[i][2] = wmma_bf16(Af[i], Bf[2], acc[i][2]);
      acc[i][3] = wmma_bf16(Af[i], Bf[3], acc[i][3]);
    }
    Af[0] = nAf0; Af[1] = nAf1;
    Bf[0] = nBf0; Bf[1] = nBf1; Bf[2] = nBf2; Bf[3] = nBf3;
  }
  #pragma unroll
  for (int i = 0; i < 2; ++i) {
    acc[i][0] = wmma_bf16(Af[i], Bf[0], acc[i][0]);
    acc[i][1] = wmma_bf16(Af[i], Bf[1], acc[i][1]);
    acc[i][2] = wmma_bf16(Af[i], Bf[2], acc[i][2]);
    acc[i][3] = wmma_bf16(Af[i], Bf[3], acc[i][3]);
  }

  #pragma unroll
  for (int i = 0; i < 2; ++i) {
    #pragma unroll
    for (int j = 0; j < 4; ++j) {
      int col = tn + j * 16 + ln;
      #pragma unroll
      for (int v = 0; v < 8; ++v) {
        int row = tm + i * 16 + v + half * 8;   // C/D layout: vgpr v -> row (+8 upper half)
        C[(size_t)row * N + col] = (OT)acc[i][j][v];
      }
    }
  }
}

// Flash attention: one wave per (16-query tile, batch, head).
__global__ void attn_kernel(const __bf16* __restrict__ qkv, __bf16* __restrict__ av) {
  __shared__ __align__(16) __bf16 lds_p[16][32];   // P tile (queries x keys)
#if defined(HAVE_ASYNC_COPY)
  __shared__ __align__(16) __bf16 lds_vr[32][64];  // V row-major (key x dim), async-DMA filled
#else
  __shared__ __align__(16) __bf16 lds_v[64][40];   // V transposed (dim x key), 80B rows
#endif
  int lane = threadIdx.x;
  int half = lane >> 4, ln = lane & 15;
  int i0 = blockIdx.x * 16;
  int b  = blockIdx.y;
  int h  = blockIdx.z;

  const __bf16* qbase = qkv + ((size_t)(M_LEN + i0 + ln) * BATCH + b) * QKV3 + h * DHEAD;
  v16bf aq0 = frag16(qbase +      (half ? 16 : 0), qbase +      (half ? 24 : 8));
  v16bf aq1 = frag16(qbase + 32 + (half ? 16 : 0), qbase + 32 + (half ? 24 : 8));

  v8f O0 = {}, O1 = {}, O2 = {}, O3 = {};
  float m_run[8], l_run[8];
  #pragma unroll
  for (int v = 0; v < 8; ++v) { m_run[v] = NEGBIG; l_run[v] = 0.f; }

  int nchunks = ((M_LEN + i0) / 16 + 2) / 2;       // 32-key chunks
  for (int c = 0; c < nchunks; ++c) {
    int j0 = c * 32;
    const __bf16* vp = qkv + ((size_t)(j0 + lane) * BATCH + b) * QKV3 + 2048 + h * DHEAD;
#if defined(HAVE_ASYNC_COPY)
    // Kick off async DMA of the V tile (32 keys x 64 dims, row-major) into LDS;
    // the score WMMAs below overlap with the copy.
    #pragma unroll
    for (int g = 0; g < 4; ++g) {
      __builtin_amdgcn_global_load_async_to_lds_b128(
          (gv4i*)(vp + g * 16),
          (lv4i*)&lds_vr[lane][g * 16],
          0, 0);
    }
#else
    #pragma unroll
    for (int g = 0; g < 8; ++g) {
      v8bf t = *(const v8bf*)(vp + g * 8);
      #pragma unroll
      for (int e = 0; e < 8; ++e) lds_v[g * 8 + e][lane] = t[e];
    }
#endif
    // Prefetch next chunk's K and V rows into cache (global_prefetch_b8).
    if (c + 1 < nchunks) {
      const __bf16* nrow = qkv + ((size_t)(j0 + 32 + lane) * BATCH + b) * QKV3;
      __builtin_prefetch(nrow + 1024 + h * DHEAD, 0, 0);   // K
      __builtin_prefetch(nrow + 2048 + h * DHEAD, 0, 0);   // V
    }
    // Scores for two 16-key sub-tiles (K-dim = DH = 64 -> 2 WMMAs each)
    v8f s0 = {}, s1 = {};
    {
      const __bf16* kp0 = qkv + ((size_t)(j0 + ln)      * BATCH + b) * QKV3 + 1024 + h * DHEAD + (half ? 16 : 0);
      const __bf16* kp1 = qkv + ((size_t)(j0 + 16 + ln) * BATCH + b) * QKV3 + 1024 + h * DHEAD + (half ? 16 : 0);
      s0 = wmma_bf16(aq0, frag16(kp0,      kp0 + 8),  s0);
      s0 = wmma_bf16(aq1, frag16(kp0 + 32, kp0 + 40), s0);
      s1 = wmma_bf16(aq0, frag16(kp1,      kp1 + 8),  s1);
      s1 = wmma_bf16(aq1, frag16(kp1 + 32, kp1 + 40), s1);
    }
    // Online softmax: rows 0-7 live in lanes 0-15, rows 8-15 in lanes 16-31.
    float osc[8];
    #pragma unroll
    for (int v = 0; v < 8; ++v) {
      int row = v + half * 8;
      float sa = s0[v] * SCALE;
      float sb = s1[v] * SCALE;
      int lim = M_LEN + i0 + row;                  // causal: key j valid iff j <= M + i
      if (j0 + ln      > lim) sa = NEGBIG;
      if (j0 + 16 + ln > lim) sb = NEGBIG;
      float t = fmaxf(sa, sb);
      t = fmaxf(t, __shfl_xor(t, 1));
      t = fmaxf(t, __shfl_xor(t, 2));
      t = fmaxf(t, __shfl_xor(t, 4));
      t = fmaxf(t, __shfl_xor(t, 8));
      float mnew = fmaxf(m_run[v], t);
      float sc = __expf(m_run[v] - mnew);
      float pa = __expf(sa - mnew);
      float pb = __expf(sb - mnew);
      float rs = pa + pb;
      rs += __shfl_xor(rs, 1);
      rs += __shfl_xor(rs, 2);
      rs += __shfl_xor(rs, 4);
      rs += __shfl_xor(rs, 8);
      l_run[v] = l_run[v] * sc + rs;
      m_run[v] = mnew;
      osc[v] = sc;
      lds_p[row][ln]      = (__bf16)pa;
      lds_p[row][16 + ln] = (__bf16)pb;
    }
    #pragma unroll
    for (int v = 0; v < 8; ++v) { O0[v] *= osc[v]; O1[v] *= osc[v]; O2[v] *= osc[v]; O3[v] *= osc[v]; }
#if defined(HAVE_ASYNC_COPY)
    wait_async0();
#endif
    __syncthreads();
    // P(16x32) x V(32x64) -> four 16x16 accumulators
    const __bf16* pr = &lds_p[ln][half ? 16 : 0];
    v16bf ap = frag16(pr, pr + 8);
    {
      const int koff = half ? 16 : 0;
#if defined(HAVE_ASYNC_COPY)
      v16bf bv0, bv1, bv2, bv3;
      #pragma unroll
      for (int e = 0; e < 16; ++e) {
        bv0[e] = lds_vr[koff + e][ 0 + ln];
        bv1[e] = lds_vr[koff + e][16 + ln];
        bv2[e] = lds_vr[koff + e][32 + ln];
        bv3[e] = lds_vr[koff + e][48 + ln];
      }
      O0 = wmma_bf16(ap, bv0, O0);
      O1 = wmma_bf16(ap, bv1, O1);
      O2 = wmma_bf16(ap, bv2, O2);
      O3 = wmma_bf16(ap, bv3, O3);
#else
      const __bf16* v0 = &lds_v[ 0 + ln][koff];
      const __bf16* v1 = &lds_v[16 + ln][koff];
      const __bf16* v2 = &lds_v[32 + ln][koff];
      const __bf16* v3 = &lds_v[48 + ln][koff];
      O0 = wmma_bf16(ap, frag16(v0, v0 + 8), O0);
      O1 = wmma_bf16(ap, frag16(v1, v1 + 8), O1);
      O2 = wmma_bf16(ap, frag16(v2, v2 + 8), O2);
      O3 = wmma_bf16(ap, frag16(v3, v3 + 8), O3);
#endif
    }
    __syncthreads();
  }
  // Normalize and store attn_vec (bf16) [S, B, H*DH]
  #pragma unroll
  for (int v = 0; v < 8; ++v) {
    int row = v + half * 8;
    float inv = 1.0f / l_run[v];
    size_t base = ((size_t)(i0 + row) * BATCH + b) * (NHEAD * DHEAD) + h * DHEAD + ln;
    av[base +  0] = (__bf16)(O0[v] * inv);
    av[base + 16] = (__bf16)(O1[v] * inv);
    av[base + 32] = (__bf16)(O2[v] * inv);
    av[base + 48] = (__bf16)(O3[v] * inv);
  }
}

// Residual + LayerNorm: one block per (s,b) row.
__global__ void ln_kernel(const float* __restrict__ x, const float* __restrict__ pos,
                          const float* __restrict__ yproj, const float* __restrict__ gamma,
                          const float* __restrict__ beta, float* __restrict__ out) {
  int row = blockIdx.x;              // s*B + b
  int s = row >> 2;
  int tid = threadIdx.x;
  __shared__ float red1[8], red2[8];
  float vals[4];
  float sum = 0.f, sumsq = 0.f;
  #pragma unroll
  for (int i = 0; i < 4; ++i) {
    int d = tid + i * 256;
    float y = x[((size_t)row << 10) + d] + pos[((size_t)s << 10) + d] + yproj[((size_t)row << 10) + d];
    vals[i] = y; sum += y; sumsq += y * y;
  }
  for (int m = 1; m < 32; m <<= 1) { sum += __shfl_xor(sum, m); sumsq += __shfl_xor(sumsq, m); }
  int wave = tid >> 5, lane = tid & 31;
  if (lane == 0) { red1[wave] = sum; red2[wave] = sumsq; }
  __syncthreads();
  if (tid == 0) {
    float a = 0.f, bb = 0.f;
    for (int w2 = 0; w2 < 8; ++w2) { a += red1[w2]; bb += red2[w2]; }
    red1[0] = a; red2[0] = bb;
  }
  __syncthreads();
  float mu  = red1[0] * (1.0f / DMODEL);
  float var = red2[0] * (1.0f / DMODEL) - mu * mu;
  float rstd = rsqrtf(var + EPS);
  #pragma unroll
  for (int i = 0; i < 4; ++i) {
    int d = tid + i * 256;
    out[((size_t)row << 10) + d] = (vals[i] - mu) * rstd * gamma[d] + beta[d];
  }
}

extern "C" void kernel_launch(void* const* d_in, const int* in_sizes, int n_in,
                              void* d_out, int out_size, void* d_ws, size_t ws_size,
                              hipStream_t stream) {
  (void)in_sizes; (void)n_in; (void)out_size; (void)ws_size;
  const float* x     = (const float*)d_in[0];
  const float* pos   = (const float*)d_in[1];
  const float* mem   = (const float*)d_in[2];
  // d_in[3] = mask : unused, causal mask is computed analytically
  const float* Wqkv  = (const float*)d_in[4];
  const float* Wo    = (const float*)d_in[5];
  const float* gamma = (const float*)d_in[6];
  const float* beta  = (const float*)d_in[7];
  float* out = (float*)d_out;

  char* w = (char*)d_ws;
  __bf16* cb    = (__bf16*)w;  w += (size_t)T_LEN * BATCH * DMODEL * 2;   // 16 MiB
  __bf16* wqkvb = (__bf16*)w;  w += (size_t)QKV3 * DMODEL * 2;            //  6 MiB
  __bf16* wob   = (__bf16*)w;  w += (size_t)DMODEL * DMODEL * 2;          //  2 MiB
  __bf16* qkvb  = (__bf16*)w;  w += (size_t)T_LEN * BATCH * QKV3 * 2;     // 48 MiB
  __bf16* avb   = (__bf16*)w;  w += (size_t)S_LEN * BATCH * DMODEL * 2;   //  8 MiB
  float*  yproj = (float*)w;                                              // 16 MiB

  pack_c_kernel<<<T_LEN * BATCH * DMODEL / 256, 256, 0, stream>>>(x, pos, mem, cb);
  f32_to_bf16_kernel<<<QKV3 * DMODEL / 256, 256, 0, stream>>>(Wqkv, wqkvb, QKV3 * DMODEL);
  f32_to_bf16_kernel<<<DMODEL * DMODEL / 256, 256, 0, stream>>>(Wo, wob, DMODEL * DMODEL);

  dim3 g1(QKV3 / 256, T_LEN * BATCH / 64);
  gemm_bf16_nt_kernel<__bf16><<<g1, 256, 0, stream>>>(cb, wqkvb, qkvb, DMODEL, QKV3);

  dim3 g2(S_LEN / 16, BATCH, NHEAD);
  attn_kernel<<<g2, 32, 0, stream>>>(qkvb, avb);

  dim3 g3(DMODEL / 256, S_LEN * BATCH / 64);
  gemm_bf16_nt_kernel<float><<<g3, 256, 0, stream>>>(avb, wob, yproj, DMODEL, DMODEL);

  ln_kernel<<<S_LEN * BATCH, 256, 0, stream>>>(x, pos, yproj, gamma, beta, out);
}
